// SimpleMAMBA_73005854097588
// MI455X (gfx1250) — matrix operand order repaired and checked by reading, not compiled, not run on hardware
//
#include <hip/hip_runtime.h>
#include <cstdint>
#include <cstddef>

typedef __attribute__((ext_vector_type(16))) __bf16 v16bf;
typedef __attribute__((ext_vector_type(8)))  __bf16 v8bf;
typedef __attribute__((ext_vector_type(8)))  float  v8f;

#define B_SZ  2
#define SEQ   2048
#define DM    1024
#define DI    2048          // d_inner
#define DSTATE 16
#define DTR   64            // dt_rank
#define MTOK  (B_SZ*SEQ)    // 4096 tokens
#define NCHNK 16
#define CLEN  128           // NCHNK*CLEN == SEQ

// ---------------------------------------------------------------- converts
__global__ __launch_bounds__(256)
void k_cvt_bf16(const float* __restrict__ s, __bf16* __restrict__ d, size_t n) {
  size_t i = (size_t)blockIdx.x * 256 + threadIdx.x;
  if (i < n) d[i] = (__bf16)s[i];
}

// pack dt_low (cols 0..63 of x_dbl, row stride 96) -> dense bf16 (MTOK x 64)
__global__ __launch_bounds__(256)
void k_pack_dtlow(const float* __restrict__ xdbl, __bf16* __restrict__ d) {
  size_t i = (size_t)blockIdx.x * 256 + threadIdx.x;
  if (i >= (size_t)MTOK * DTR) return;
  size_t r = i >> 6; int c = (int)(i & 63);
  d[i] = (__bf16)xdbl[r * 96 + c];
}

// ---------------------------------------------------------------- WMMA GEMM
// C(MxN) = A(MxK,bf16) * W(NxK,bf16)^T  [+bias] [+activation]
// block: 256 thr = 8 waves; tile 128x128x32; wave -> 32x64 (8x 16x16 wmma)
// LDS double-buffered (ping-pong): one barrier per K-step, global loads for
// step k+1 issued before the WMMA block of step k.
enum { EPI_F32 = 0, EPI_F32_BIAS = 1, EPI_BF16_BIAS = 2, EPI_BF16 = 3,
       EPI_F32_SOFTPLUS_BIAS = 4 };

template<int EPI>
__global__ __launch_bounds__(256)
void k_gemm_bf16(const __bf16* __restrict__ A, const __bf16* __restrict__ W,
                 const float* __restrict__ bias, void* __restrict__ outp,
                 int M, int N, int K) {
  constexpr int BM = 128, BN = 128, BK = 32;
  constexpr int LDT = BK + 8;                    // 80B row stride: conflict-free b128
  __shared__ __bf16 sA[2][BM * LDT];             // 2 x 10 KB
  __shared__ __bf16 sB[2][BN * LDT];             // 2 x 10 KB

  const int tid  = threadIdx.x;
  const int lane = tid & 31, wave = tid >> 5;
  const int wm = wave & 3, wn = wave >> 2;       // 4x2 wave grid -> 32x64 per wave
  const int bm0 = blockIdx.x * BM, bn0 = blockIdx.y * BN;
  const int l15 = lane & 15, lhi = lane >> 4;

  const int ldRow = tid >> 1, ldCol = (tid & 1) * 16;  // 128 rows x 2 halves (32B)

  v8f acc[2][4] = {{{}, {}, {}, {}}, {{}, {}, {}, {}}};

  // ---- preload K-step 0 into buffer 0
  {
    const __bf16* g = A + (size_t)(bm0 + ldRow) * K + ldCol;
    v8bf a0 = *(const v8bf*)g;
    v8bf a1 = *(const v8bf*)(g + 8);
    *(v8bf*)&sA[0][ldRow * LDT + ldCol]     = a0;
    *(v8bf*)&sA[0][ldRow * LDT + ldCol + 8] = a1;
    v8bf b0 = {}, b1 = {};
    int gr = bn0 + ldRow;
    if (gr < N) {
      const __bf16* gw = W + (size_t)gr * K + ldCol;
      b0 = *(const v8bf*)gw;
      b1 = *(const v8bf*)(gw + 8);
    }
    *(v8bf*)&sB[0][ldRow * LDT + ldCol]     = b0;
    *(v8bf*)&sB[0][ldRow * LDT + ldCol + 8] = b1;
  }
  __syncthreads();

  for (int kk = 0; kk < K; kk += BK) {
    const int cur = (kk / BK) & 1, nxt = cur ^ 1;
    const bool haveNext = (kk + BK) < K;

    // ---- issue next-tile global loads first (retire after the WMMA block)
    v8bf na0 = {}, na1 = {}, nb0 = {}, nb1 = {};
    if (haveNext) {
      const __bf16* g = A + (size_t)(bm0 + ldRow) * K + kk + BK + ldCol;
      na0 = *(const v8bf*)g;
      na1 = *(const v8bf*)(g + 8);
      int gr = bn0 + ldRow;
      if (gr < N) {
        const __bf16* gw = W + (size_t)gr * K + kk + BK + ldCol;
        nb0 = *(const v8bf*)gw;
        nb1 = *(const v8bf*)(gw + 8);
      }
      if (kk + 2 * BK < K)  // hint K-step k+2 into cache (global_prefetch_b8)
        __builtin_prefetch(A + (size_t)(bm0 + ldRow) * K + kk + 2 * BK + ldCol, 0, 1);
    }

    // ---- fragments + WMMA from current buffer (ISA 16-bit layouts)
    const int akb = lhi * 8, bkb = lhi * 8;
    v16bf af[2], bfr[4];
#pragma unroll
    for (int mi = 0; mi < 2; ++mi) {
      const int r = wm * 32 + mi * 16 + l15;
      v8bf x0 = *(const v8bf*)&sA[cur][r * LDT + akb];
      v8bf x1 = *(const v8bf*)&sA[cur][r * LDT + akb + 16];
#pragma unroll
      for (int i = 0; i < 8; ++i) { af[mi][i] = x0[i]; af[mi][i + 8] = x1[i]; }
    }
#pragma unroll
    for (int ni = 0; ni < 4; ++ni) {
      const int c = wn * 64 + ni * 16 + l15;
      v8bf y0 = *(const v8bf*)&sB[cur][c * LDT + bkb];
      v8bf y1 = *(const v8bf*)&sB[cur][c * LDT + bkb + 16];
#pragma unroll
      for (int i = 0; i < 8; ++i) { bfr[ni][i] = y0[i]; bfr[ni][i + 8] = y1[i]; }
    }
#pragma unroll
    for (int mi = 0; mi < 2; ++mi)
#pragma unroll
      for (int ni = 0; ni < 4; ++ni)
        acc[mi][ni] = __builtin_amdgcn_wmma_f32_16x16x32_bf16(
            false, af[mi], false, bfr[ni], (short)0, acc[mi][ni], false, false);

    // ---- publish next buffer, single barrier per K-step
    if (haveNext) {
      *(v8bf*)&sA[nxt][ldRow * LDT + ldCol]     = na0;
      *(v8bf*)&sA[nxt][ldRow * LDT + ldCol + 8] = na1;
      *(v8bf*)&sB[nxt][ldRow * LDT + ldCol]     = nb0;
      *(v8bf*)&sB[nxt][ldRow * LDT + ldCol + 8] = nb1;
    }
    __syncthreads();
  }

  // epilogue: C layout VGPR v -> row (lhi*8+v), col l15
#pragma unroll
  for (int mi = 0; mi < 2; ++mi)
#pragma unroll
    for (int ni = 0; ni < 4; ++ni) {
      const int gc = bn0 + wn * 64 + ni * 16 + l15;
      if (gc >= N) continue;
      float bv = 0.f;
      if (EPI == EPI_F32_BIAS || EPI == EPI_BF16_BIAS || EPI == EPI_F32_SOFTPLUS_BIAS)
        bv = bias[gc];
#pragma unroll
      for (int v = 0; v < 8; ++v) {
        const int gr = bm0 + wm * 32 + mi * 16 + lhi * 8 + v;
        float c = acc[mi][ni][v] + bv;
        size_t idx = (size_t)gr * N + gc;
        if (EPI == EPI_F32 || EPI == EPI_F32_BIAS) {
          ((float*)outp)[idx] = c;
        } else if (EPI == EPI_F32_SOFTPLUS_BIAS) {
          ((float*)outp)[idx] = fmaxf(c, 0.f) + log1pf(__expf(-fabsf(c)));
        } else {
          ((__bf16*)outp)[idx] = (__bf16)c;
        }
      }
    }
}

// ---------------------------------------------------------------- conv+SiLU
__global__ __launch_bounds__(256)
void k_conv_silu(const float* __restrict__ xz, const float* __restrict__ cw,
                 const float* __restrict__ cb, __bf16* __restrict__ u) {
  size_t i = (size_t)blockIdx.x * 256 + threadIdx.x;
  if (i >= (size_t)MTOK * DI) return;
  int d = (int)(i & (DI - 1));
  size_t row = i >> 11;                 // DI == 2048
  int t = (int)(row & (SEQ - 1));
  size_t brow0 = row - t;               // b*SEQ
  float s = cb[d];
#pragma unroll
  for (int w = 0; w < 4; ++w) {
    int tt = t - 3 + w;
    if (tt >= 0) s += xz[(brow0 + tt) * (size_t)(2 * DI) + d] * cw[d * 4 + w];
  }
  u[i] = (__bf16)(s / (1.f + __expf(-s)));
}

// ---------------------------------------------------------------- scan
// pass1: per-chunk suffix accumulator Q[n] and sum(dt)  (h_out = exp(A*sumdt)*h_in + Q)
__global__ __launch_bounds__(256)
void k_scan_pass1(const float* __restrict__ dt, const __bf16* __restrict__ u,
                  const float* __restrict__ xdbl, const float* __restrict__ A_log,
                  float* __restrict__ Q, float* __restrict__ sumdt) {
  const int d = blockIdx.x * 256 + threadIdx.x;   // 0..DI-1
  const int ch = blockIdx.y, b = blockIdx.z;
  __shared__ float sBm[CLEN * DSTATE];            // 8 KB
  float Arow[DSTATE];
#pragma unroll
  for (int n = 0; n < DSTATE; ++n) Arow[n] = -__expf(A_log[d * DSTATE + n]);
  // cooperative stage of Bm (cols 64..79 of x_dbl) for this chunk
  const int t0 = ch * CLEN;
  for (int i = threadIdx.x; i < CLEN * DSTATE; i += 256) {
    int r = i >> 4, c = i & 15;
    sBm[i] = xdbl[((size_t)(b * SEQ + t0 + r)) * 96 + 64 + c];
  }
  __syncthreads();
  float q[DSTATE] = {};
  float sd = 0.f;
  for (int t = 0; t < CLEN; ++t) {
    size_t gi = ((size_t)(b * SEQ + t0 + t)) * DI + d;
    float dtv = dt[gi];
    float du = dtv * (float)u[gi];
    sd += dtv;
#pragma unroll
    for (int n = 0; n < DSTATE; ++n) {
      float dA = __expf(dtv * Arow[n]);
      q[n] = dA * q[n] + du * sBm[t * DSTATE + n];
    }
  }
  size_t qi = (((size_t)b * NCHNK + ch) * DI + d) * DSTATE;
#pragma unroll
  for (int n = 0; n < DSTATE; ++n) Q[qi + n] = q[n];
  sumdt[((size_t)b * NCHNK + ch) * DI + d] = sd;
}

// mid: sequential combine over 16 chunks -> h0 per chunk
__global__ __launch_bounds__(256)
void k_scan_mid(const float* __restrict__ Q, const float* __restrict__ sumdt,
                const float* __restrict__ A_log, float* __restrict__ h0) {
  const int d = blockIdx.x * 256 + threadIdx.x;
  const int b = blockIdx.y;
  float Arow[DSTATE];
#pragma unroll
  for (int n = 0; n < DSTATE; ++n) Arow[n] = -__expf(A_log[d * DSTATE + n]);
  float h[DSTATE] = {};
  for (int c = 0; c < NCHNK; ++c) {
    size_t base = (((size_t)b * NCHNK + c) * DI + d) * DSTATE;
    float sd = sumdt[((size_t)b * NCHNK + c) * DI + d];
#pragma unroll
    for (int n = 0; n < DSTATE; ++n) {
      h0[base + n] = h[n];
      h[n] = __expf(Arow[n] * sd) * h[n] + Q[base + n];
    }
  }
}

// pass2: replay chunk from correct h0; fuse y = h.C + D*u, gate by silu(z), emit bf16
__global__ __launch_bounds__(256)
void k_scan_pass2(const float* __restrict__ dt, const __bf16* __restrict__ u,
                  const float* __restrict__ xdbl, const float* __restrict__ A_log,
                  const float* __restrict__ Dp, const float* __restrict__ xz,
                  const float* __restrict__ h0, __bf16* __restrict__ yg) {
  const int d = blockIdx.x * 256 + threadIdx.x;
  const int ch = blockIdx.y, b = blockIdx.z;
  __shared__ float sBC[CLEN * 32];                // 16 KB: Bm then Cm per step
  float Arow[DSTATE];
#pragma unroll
  for (int n = 0; n < DSTATE; ++n) Arow[n] = -__expf(A_log[d * DSTATE + n]);
  const float Dv = Dp[d];
  const int t0 = ch * CLEN;
  for (int i = threadIdx.x; i < CLEN * 32; i += 256) {
    int r = i >> 5, c = i & 31;
    sBC[i] = xdbl[((size_t)(b * SEQ + t0 + r)) * 96 + 64 + c];
  }
  __syncthreads();
  float h[DSTATE];
  size_t hb = (((size_t)b * NCHNK + ch) * DI + d) * DSTATE;
#pragma unroll
  for (int n = 0; n < DSTATE; ++n) h[n] = h0[hb + n];
  for (int t = 0; t < CLEN; ++t) {
    size_t row = (size_t)(b * SEQ + t0 + t);
    size_t gi = row * DI + d;
    float dtv = dt[gi];
    float uv = (float)u[gi];
    float du = dtv * uv;
    float y = 0.f;
#pragma unroll
    for (int n = 0; n < DSTATE; ++n) {
      float dA = __expf(dtv * Arow[n]);
      h[n] = dA * h[n] + du * sBC[t * 32 + n];
      y += h[n] * sBC[t * 32 + 16 + n];
    }
    y += Dv * uv;
    float zv = xz[row * (size_t)(2 * DI) + DI + d];
    yg[gi] = (__bf16)(y * zv / (1.f + __expf(-zv)));
  }
}

// ---------------------------------------------------------------- launch
extern "C" void kernel_launch(void* const* d_in, const int* in_sizes, int n_in,
                              void* d_out, int out_size, void* d_ws, size_t ws_size,
                              hipStream_t stream) {
  (void)in_sizes; (void)n_in; (void)out_size; (void)ws_size;
  const float* x         = (const float*)d_in[0];
  const float* fc1_w     = (const float*)d_in[1];
  const float* fc1_b     = (const float*)d_in[2];
  const float* in_proj_w = (const float*)d_in[3];
  const float* conv_w    = (const float*)d_in[4];
  const float* conv_b    = (const float*)d_in[5];
  const float* x_proj_w  = (const float*)d_in[6];
  const float* dt_proj_w = (const float*)d_in[7];
  const float* dt_proj_b = (const float*)d_in[8];
  const float* A_log     = (const float*)d_in[9];
  const float* Dp        = (const float*)d_in[10];
  const float* out_proj_w= (const float*)d_in[11];
  const float* fc2_w     = (const float*)d_in[12];
  const float* fc2_b     = (const float*)d_in[13];
  float* out = (float*)d_out;

  char* ws = (char*)d_ws;
  size_t off = 0;
  auto take = [&](size_t bytes) { size_t o = off; off += (bytes + 255) & ~(size_t)255; return o; };
  __bf16* wFc1   = (__bf16*)(ws + take((size_t)DM * DM * 2));
  __bf16* wInp   = (__bf16*)(ws + take((size_t)2 * DI * DM * 2));
  __bf16* wXp    = (__bf16*)(ws + take((size_t)96 * DI * 2));
  __bf16* wDtp   = (__bf16*)(ws + take((size_t)DI * DTR * 2));
  __bf16* wOutp  = (__bf16*)(ws + take((size_t)DM * DI * 2));
  __bf16* wFc2   = (__bf16*)(ws + take((size_t)DM * DM * 2));
  __bf16* xBf    = (__bf16*)(ws + take((size_t)MTOK * DM * 2));
  __bf16* hBf    = (__bf16*)(ws + take((size_t)MTOK * DM * 2));
  float*  xzF    = (float*) (ws + take((size_t)MTOK * 2 * DI * 4));
  __bf16* uBf    = (__bf16*)(ws + take((size_t)MTOK * DI * 2));
  float*  xdblF  = (float*) (ws + take((size_t)MTOK * 96 * 4));
  __bf16* dtlBf  = (__bf16*)(ws + take((size_t)MTOK * DTR * 2));
  float*  dtF    = (float*) (ws + take((size_t)MTOK * DI * 4));
  float*  Qf     = (float*) (ws + take((size_t)B_SZ * NCHNK * DI * DSTATE * 4));
  float*  sdF    = (float*) (ws + take((size_t)B_SZ * NCHNK * DI * 4));
  float*  h0F    = (float*) (ws + take((size_t)B_SZ * NCHNK * DI * DSTATE * 4));
  __bf16* ygBf   = (__bf16*)(ws + take((size_t)MTOK * DI * 2));
  __bf16* o1Bf   = (__bf16*)(ws + take((size_t)MTOK * DM * 2));

  dim3 blk(256);
  auto nb = [](size_t n) { return (unsigned)((n + 255) / 256); };

  // weight + input down-converts (one-time per call)
  k_cvt_bf16<<<nb((size_t)DM*DM), blk, 0, stream>>>(fc1_w, wFc1, (size_t)DM*DM);
  k_cvt_bf16<<<nb((size_t)2*DI*DM), blk, 0, stream>>>(in_proj_w, wInp, (size_t)2*DI*DM);
  k_cvt_bf16<<<nb((size_t)96*DI), blk, 0, stream>>>(x_proj_w, wXp, (size_t)96*DI);
  k_cvt_bf16<<<nb((size_t)DI*DTR), blk, 0, stream>>>(dt_proj_w, wDtp, (size_t)DI*DTR);
  k_cvt_bf16<<<nb((size_t)DM*DI), blk, 0, stream>>>(out_proj_w, wOutp, (size_t)DM*DI);
  k_cvt_bf16<<<nb((size_t)DM*DM), blk, 0, stream>>>(fc2_w, wFc2, (size_t)DM*DM);
  k_cvt_bf16<<<nb((size_t)MTOK*DM), blk, 0, stream>>>(x, xBf, (size_t)MTOK*DM);

  // fc1: h = x @ fc1_w^T + b  -> bf16
  k_gemm_bf16<EPI_BF16_BIAS><<<dim3(MTOK/128, DM/128), blk, 0, stream>>>(
      xBf, wFc1, fc1_b, hBf, MTOK, DM, DM);
  // in_proj: xz = h @ in_proj_w^T -> f32
  k_gemm_bf16<EPI_F32><<<dim3(MTOK/128, (2*DI)/128), blk, 0, stream>>>(
      hBf, wInp, nullptr, xzF, MTOK, 2*DI, DM);
  // causal depthwise conv + SiLU -> u (bf16)
  k_conv_silu<<<nb((size_t)MTOK*DI), blk, 0, stream>>>(xzF, conv_w, conv_b, uBf);
  // x_proj: x_dbl = u @ x_proj_w^T -> f32 (N=96, bounds-checked)
  k_gemm_bf16<EPI_F32><<<dim3(MTOK/128, 1), blk, 0, stream>>>(
      uBf, wXp, nullptr, xdblF, MTOK, 96, DI);
  // dt = softplus(dt_low @ dt_proj_w^T + dt_proj_b) -> f32
  k_pack_dtlow<<<nb((size_t)MTOK*DTR), blk, 0, stream>>>(xdblF, dtlBf);
  k_gemm_bf16<EPI_F32_SOFTPLUS_BIAS><<<dim3(MTOK/128, DI/128), blk, 0, stream>>>(
      dtlBf, wDtp, dt_proj_b, dtF, MTOK, DI, DTR);
  // chunked selective scan (3 phases) + fused D*u skip and silu(z) gating
  k_scan_pass1<<<dim3(DI/256, NCHNK, B_SZ), blk, 0, stream>>>(dtF, uBf, xdblF, A_log, Qf, sdF);
  k_scan_mid<<<dim3(DI/256, B_SZ), blk, 0, stream>>>(Qf, sdF, A_log, h0F);
  k_scan_pass2<<<dim3(DI/256, NCHNK, B_SZ), blk, 0, stream>>>(
      dtF, uBf, xdblF, A_log, Dp, xzF, h0F, ygBf);
  // out_proj -> bf16
  k_gemm_bf16<EPI_BF16><<<dim3(MTOK/128, DM/128), blk, 0, stream>>>(
      ygBf, wOutp, nullptr, o1Bf, MTOK, DM, DI);
  // fc2 + bias -> f32 output
  k_gemm_bf16<EPI_F32_BIAS><<<dim3(MTOK/128, DM/128), blk, 0, stream>>>(
      o1Bf, wFc2, fc2_b, out, MTOK, DM, DM);
}
